// SimplePixelRegressor_19567871001331
// MI455X (gfx1250) — matrix operand order repaired and checked by reading, not compiled, not run on hardware
//
#include <hip/hip_runtime.h>

typedef __attribute__((ext_vector_type(16))) _Float16 v16h;
typedef __attribute__((ext_vector_type(8)))  _Float16 v8h;
typedef __attribute__((ext_vector_type(8)))  float    v8f;

static constexpr int kH = 32;        // hidden
static constexpr int kT = 1024;      // timesteps
static constexpr int kB = 4096;      // batch
static constexpr int kWaves = 4;     // waves per block
static constexpr int kBlock = kWaves * 32;
static constexpr int kNWaves = kB / 16;          // 256 waves total
static constexpr float kInvN = 1.0f / (float)(kB * kT * 3);

// fast activations: v_exp_f32 + v_rcp_f32 (1 ulp), no IEEE div sequence
__device__ __forceinline__ float sig_f(float x) {
  return __builtin_amdgcn_rcpf(1.0f + __expf(-x));
}
__device__ __forceinline__ float tanh_f(float x) {
  return __builtin_fmaf(2.0f, __builtin_amdgcn_rcpf(1.0f + __expf(-2.0f * x)), -1.0f);
}

extern "C" __global__ __launch_bounds__(kBlock)
void gru_fused_kernel(const float* __restrict__ X,     // [B,T,3]
                      const float* __restrict__ TGT,   // [B,T,3]
                      const float* __restrict__ W_ih,  // [96,3]
                      const float* __restrict__ W_hh,  // [96,32]
                      const float* __restrict__ b_ih,  // [96]
                      const float* __restrict__ b_hh,  // [96]
                      const float* __restrict__ fc_W,  // [3,32]
                      const float* __restrict__ fc_b,  // [3]
                      float* __restrict__ out,         // [1 + B*T*3]
                      float* __restrict__ ws)          // >= kNWaves floats
{
  __shared__ __align__(16) _Float16 h16s[kWaves][16 * kH]; // [row][col] f16 h
  __shared__ __align__(16) float    xs[kWaves][16][4];     // staged x rows

  const int tid  = threadIdx.x;
  const int w    = tid >> 5;
  const int lane = tid & 31;
  const int l16  = lane & 15;
  const int hih  = lane >> 4;            // 0: lanes 0-15, 1: lanes 16-31
  const int moff = hih * 8;              // D-layout row offset for this lane half
  const int wid  = blockIdx.x * kWaves + w;
  const int b0   = wid * 16;             // first batch row of this wave

  _Float16* hl = h16s[w];

  // ---- Pack W_hh into 6 f16 WMMA B operands (gate r/z/n  x  N-tile 0/1) ----
  // B k order used here (contiguous, same as hA): VGPR v, lane L -> (K = 2v + (L<16?0:16) + j, N = L%16)
  v16h Bg[3][2];
#pragma unroll
  for (int g = 0; g < 3; ++g)
#pragma unroll
    for (int nt = 0; nt < 2; ++nt) {
      const int n = nt * 16 + l16;
      const float* wrow = W_hh + (g * kH + n) * kH;  // result[m][n]=sum_k h[m][k]*W[n][k]
#pragma unroll
      for (int v = 0; v < 8; ++v)
#pragma unroll
        for (int j = 0; j < 2; ++j) {
          const int k = 2 * v + hih * 16 + j;
          Bg[g][nt][2 * v + j] = (_Float16)wrow[k];
        }
    }
  // fc head as a B operand: only N=0..2 columns are live
  v16h Bfc;
#pragma unroll
  for (int v = 0; v < 8; ++v)
#pragma unroll
    for (int j = 0; j < 2; ++j) {
      const int k = 2 * v + hih * 16 + j;
      Bfc[2 * v + j] = (l16 < 3) ? (_Float16)fc_W[l16 * kH + k] : (_Float16)0.0f;
    }

  // ---- Per-lane input-gate weights / fused biases for its two h-columns ----
  float wih[2][3][3], bias_r[2], bias_z[2], bias_nx[2];
  v8f bnhC[2];   // bias_nh splat, fed as WMMA C for the n gate
#pragma unroll
  for (int nt = 0; nt < 2; ++nt) {
    const int c = nt * 16 + l16;
#pragma unroll
    for (int g = 0; g < 3; ++g)
#pragma unroll
      for (int i = 0; i < 3; ++i)
        wih[nt][g][i] = W_ih[(g * kH + c) * 3 + i];
    bias_r[nt]  = b_ih[c]        + b_hh[c];
    bias_z[nt]  = b_ih[kH + c]   + b_hh[kH + c];
    bias_nx[nt] = b_ih[2*kH + c];
    const float bnh = b_hh[2*kH + c];
#pragma unroll
    for (int g = 0; g < 8; ++g) bnhC[nt][g] = bnh;
  }
  v8f Cfc;       // fc bias splat, fed as WMMA C for the head
  {
    const float fcb = (l16 < 3) ? fc_b[l16] : 0.0f;
#pragma unroll
    for (int g = 0; g < 8; ++g) Cfc[g] = fcb;
  }

  // ---- State: f32 master h in D layout [tile][row], f16 A-layout copy ----
  float h[2][8];
#pragma unroll
  for (int nt = 0; nt < 2; ++nt)
#pragma unroll
    for (int g = 0; g < 8; ++g) h[nt][g] = 0.0f;
  v16h hA = {};          // h0 = 0
  float acc = 0.0f;

  for (int t = 0; t < kT; ++t) {
    // stage x[t] for the wave's 16 rows into LDS (same-wave DS is in-order)
    if (hih == 0) {
      const float* xp = X + ((size_t)(b0 + lane) * kT + t) * 3;
      xs[w][lane][0] = xp[0];
      xs[w][lane][1] = xp[1];
      xs[w][lane][2] = xp[2];
    }

#pragma unroll
    for (int nt = 0; nt < 2; ++nt) {
      // x-side gate values (include b_ih, and b_hh for r/z) -> WMMA C operands
      v8f xr, xz, xn;
#pragma unroll
      for (int g = 0; g < 8; ++g) {
        const int m = g + moff;
        const float x0 = xs[w][m][0], x1 = xs[w][m][1], x2 = xs[w][m][2];
        xr[g] = bias_r[nt]  + wih[nt][0][0]*x0 + wih[nt][0][1]*x1 + wih[nt][0][2]*x2;
        xz[g] = bias_z[nt]  + wih[nt][1][0]*x0 + wih[nt][1][1]*x1 + wih[nt][1][2]*x2;
        xn[g] = bias_nx[nt] + wih[nt][2][0]*x0 + wih[nt][2][1]*x1 + wih[nt][2][2]*x2;
      }
      // recurrent matvecs: D = hA(16x32 f16) x W^T(32x16 f16) + C, f32 accum
      v8f rp = __builtin_amdgcn_wmma_f32_16x16x32_f16(false, hA, false, Bg[0][nt], (short)0, xr,       false, false);
      v8f zp = __builtin_amdgcn_wmma_f32_16x16x32_f16(false, hA, false, Bg[1][nt], (short)0, xz,       false, false);
      v8f np = __builtin_amdgcn_wmma_f32_16x16x32_f16(false, hA, false, Bg[2][nt], (short)0, bnhC[nt], false, false);
#pragma unroll
      for (int g = 0; g < 8; ++g) {
        const int m = g + moff;
        const float r  = sig_f(rp[g]);
        const float z  = sig_f(zp[g]);
        const float n  = tanh_f(xn[g] + r * np[g]);
        const float hn = (1.0f - z) * n + z * h[nt][g];
        h[nt][g] = hn;
        // D-layout scatter into [row][col] f16 LDS tile
        hl[m * kH + (nt * 16 + l16)] = (_Float16)hn;
      }
    }

    // rebuild A-layout f16 h for next step (and for fc WMMA):
    // lane L: row M=L%16, K = {0..15} (L<16) or {16..31} (L>=16)
    {
      const int base = l16 * kH + hih * 16;
      const v8h lo = *(const v8h*)(hl + base);
      const v8h hi = *(const v8h*)(hl + base + 8);
#pragma unroll
      for (int i = 0; i < 8; ++i) { hA[i] = lo[i]; hA[8 + i] = hi[i]; }
    }

    // fc head: preds = h x fc_W^T + fc_b via one WMMA (cols 0..2 live)
    v8f pd = __builtin_amdgcn_wmma_f32_16x16x32_f16(false, hA, false, Bfc, (short)0, Cfc, false, false);
    if (l16 < 3) {
      const int o = l16;
#pragma unroll
      for (int g = 0; g < 8; ++g) {
        const int m = g + moff;
        const size_t idx = ((size_t)(b0 + m) * kT + t) * 3 + o;
        const float p = pd[g];
        out[1 + idx] = p;
        const float d = p - TGT[idx];
        acc += d * d;
      }
    }
  }

  // wave-level loss reduction -> one workspace slot per wave (deterministic)
#pragma unroll
  for (int off = 16; off > 0; off >>= 1)
    acc += __shfl_down(acc, off, 32);
  if (lane == 0) ws[wid] = acc;
}

extern "C" __global__ __launch_bounds__(256)
void loss_reduce_kernel(const float* __restrict__ ws, float* __restrict__ out) {
  __shared__ float s[256];
  const int i = threadIdx.x;
  s[i] = (i < kNWaves) ? ws[i] : 0.0f;
  __syncthreads();
#pragma unroll
  for (int off = 128; off > 0; off >>= 1) {
    if (i < off) s[i] += s[i + off];
    __syncthreads();
  }
  if (i == 0) out[0] = s[0] * kInvN;
}

extern "C" void kernel_launch(void* const* d_in, const int* in_sizes, int n_in,
                              void* d_out, int out_size, void* d_ws, size_t ws_size,
                              hipStream_t stream) {
  const float* X    = (const float*)d_in[0];
  const float* TGT  = (const float*)d_in[1];
  const float* W_ih = (const float*)d_in[2];
  const float* W_hh = (const float*)d_in[3];
  const float* b_ih = (const float*)d_in[4];
  const float* b_hh = (const float*)d_in[5];
  const float* fc_W = (const float*)d_in[6];
  const float* fc_b = (const float*)d_in[7];
  float* out = (float*)d_out;
  float* ws  = (float*)d_ws;

  const int grid = kB / (kWaves * 16);  // 64 blocks, 16 batch rows per wave
  gru_fused_kernel<<<grid, kBlock, 0, stream>>>(X, TGT, W_ih, W_hh, b_ih, b_hh,
                                                fc_W, fc_b, out, ws);
  loss_reduce_kernel<<<1, 256, 0, stream>>>(ws, out);
}
